// HypergraphProteinRegressionModel_816043786316
// MI455X (gfx1250) — compile-verified, matrix operand order checked by reading, not run on hardware
//
#include <hip/hip_runtime.h>
#include <hip/hip_bf16.h>

// ---------------------------------------------------------------------------
// Problem constants (match reference)
// ---------------------------------------------------------------------------
#define N_P   100000
#define N_HE  20000
#define N_INC 400000
#define DD    256
#define NHH   4
#define HDD   64

typedef __bf16 bf16_t;
typedef __attribute__((ext_vector_type(16))) bf16_t v16bf;
typedef __attribute__((ext_vector_type(8)))  float  v8f;

union FragAB { v16bf v; unsigned short u[16]; };
union FragC  { v8f   v; float          f[8];  };

// float -> bf16 bits via hardware convert
static __device__ __forceinline__ unsigned short f2bfu(float f) {
    return __builtin_bit_cast(unsigned short, (bf16_t)f);
}

// K-offset inside a 16-bit 16x32 A (or 32x16 B) fragment:
// element e = {vgpr j = e>>1, half hb = e&1}; lanes 16-31 shift K by +8.
static __device__ __forceinline__ int kmap(int e, int laneHi) {
    int j = e >> 1, hb = e & 1;
    int base = (j < 4) ? (j << 1) : (16 + ((j - 4) << 1));
    return base + hb + (laneHi ? 8 : 0);
}

// inverse of kmap: given k-in-32, produce (lane-hi, element) pair
static __device__ __forceinline__ void kmap_inv(int kk, int& hi, int& e) {
    hi = (kk >> 3) & 1;
    int base = (kk < 16) ? (kk - (hi ? 8 : 0)) : (kk - 16 - (hi ? 8 : 0)); // 0..7
    int j = (kk < 16) ? (base >> 1) : (4 + (base >> 1));
    e = (j << 1) | (kk & 1);
}

static __device__ __forceinline__ v8f wmma_bf16(v16bf a, v16bf b, v8f c) {
    return __builtin_amdgcn_wmma_f32_16x16x32_bf16(
        /*neg_a=*/false, a, /*neg_b=*/false, b,
        /*c_mod=*/(short)0, c, /*reuse_a=*/false, /*reuse_b=*/false);
}

// contiguous 32-byte fragment fetch (2x b128)
static __device__ __forceinline__ v16bf frag_ld(const unsigned short* p) {
    return *(const v16bf*)p;
}

// ---------------------------------------------------------------------------
// Weight prep: W (N rows x K cols, f32, row-major) -> bf16 fragments of W^T
// Layout: [nt = n/16][ks = k/32][lane 0..31][e 0..15], 512 ushorts per (nt,ks)
// One thread per (nt,ks,lane); writes one contiguous 32-byte fragment.
// ---------------------------------------------------------------------------
__global__ void swizzle_wT(const float* __restrict__ src,
                           unsigned short* __restrict__ dst,
                           int N, int K) {
    int tid = blockIdx.x * blockDim.x + threadIdx.x;
    int total = (N >> 4) * (K >> 5) * 32;
    if (tid >= total) return;
    int lane = tid & 31;
    int rest = tid >> 5;
    int ks = rest % (K >> 5);
    int nt = rest / (K >> 5);
    int n  = (nt << 4) + (lane & 15);
    int hi = lane >> 4;
    FragAB fa;
#pragma unroll
    for (int e = 0; e < 16; ++e) {
        int k = (ks << 5) + kmap(e, hi);
        fa.u[e] = f2bfu(src[(size_t)n * K + k]);
    }
    *(v16bf*)(dst + (size_t)tid * 16) = fa.v;
}

// Wh1 (NH, D, HD) -> fragments of B[d][h*64+kk]  (K=256, N=256)
__global__ void swizzle_wh1(const float* __restrict__ src,
                            unsigned short* __restrict__ dst) {
    int tid = blockIdx.x * blockDim.x + threadIdx.x;
    if (tid >= 16 * 8 * 32) return;       // nt(16) * ks(8) * lane(32)
    int lane = tid & 31;
    int rest = tid >> 5;
    int ks = rest & 7;
    int nt = rest >> 3;
    int n  = (nt << 4) + (lane & 15);     // column c = h*64+kk
    int h  = n >> 6, kk = n & 63;
    int hi = lane >> 4;
    FragAB fa;
#pragma unroll
    for (int e = 0; e < 16; ++e) {
        int d = (ks << 5) + kmap(e, hi);  // K index = d
        fa.u[e] = f2bfu(src[h * (DD * HDD) + d * HDD + kk]);
    }
    *(v16bf*)(dst + (size_t)tid * 16) = fa.v;
}

// ---------------------------------------------------------------------------
// Weighted scatter-add:  dst[dst_idx[i]] += src[src_idx[i]] * ew[i]
// ---------------------------------------------------------------------------
__global__ void scatter_weighted_add(const float* __restrict__ src,
                                     const int*   __restrict__ src_idx,
                                     const int*   __restrict__ dst_idx,
                                     const float* __restrict__ ew,
                                     float*       __restrict__ dst,
                                     int n_inc) {
    int i = blockIdx.x * 4 + (threadIdx.x >> 6);
    if (i >= n_inc) return;
    int lane4 = (threadIdx.x & 63) << 2;
    float w = ew[i];
    const float* s = src + (size_t)src_idx[i] * DD + lane4;
    float*       d = dst + (size_t)dst_idx[i] * DD + lane4;
    float4 v = *reinterpret_cast<const float4*>(s);
    unsafeAtomicAdd(d + 0, v.x * w);
    unsafeAtomicAdd(d + 1, v.y * w);
    unsafeAtomicAdd(d + 2, v.z * w);
    unsafeAtomicAdd(d + 3, v.w * w);
}

// ---------------------------------------------------------------------------
// Stage a 16xK f32 tile into LDS as bf16 fragments (fragment-ordered).
// blockDim = 256; thread t handles pair index t + p*256 -> (ks, lane).
// ---------------------------------------------------------------------------
static __device__ __forceinline__ void stage_a_tile(const float* __restrict__ A,
                                                    int K, int rowbase,
                                                    unsigned short* sA) {
    int t = threadIdx.x;
    int npairs = (K >> 5) * 32;           // (K/32) k-steps * 32 lanes
    for (int pi = t; pi < npairs; pi += 256) {
        int lane = pi & 31;
        int ks   = pi >> 5;
        int row  = rowbase + (lane & 15);
        int hi   = lane >> 4;
        FragAB fa;
#pragma unroll
        for (int e = 0; e < 16; ++e) {
            int k = (ks << 5) + kmap(e, hi);
            fa.u[e] = f2bfu(A[(size_t)row * K + k]);
        }
        *(v16bf*)(sA + pi * 16) = fa.v;
    }
}

// ---------------------------------------------------------------------------
// Hyperedge attention (fused): h1 = relu(heF @ Bh1 + bh1);
// attn = sigmoid(h1 . Wh2 + bh2) @ Wf;  heF *= attn   (in place)
// block = 256 thr (8 waves), 16 hyperedges per block, N = 256, K = 256
// ---------------------------------------------------------------------------
__global__ void he_attn_kernel(float* __restrict__ heF,
                               const unsigned short* __restrict__ Bh1,
                               const float* __restrict__ bh1,
                               const float* __restrict__ Wh2,
                               const float* __restrict__ bh2,
                               const float* __restrict__ Wf) {
    __shared__ __align__(32) unsigned short sA[8 * 512];   // 8 KB
    __shared__ float part[16 * NHH];
    __shared__ float attn[16];

    int t = threadIdx.x;
    if (t < 16 * NHH) part[t] = 0.0f;

    int rowbase = blockIdx.x * 16;
    stage_a_tile(heF, 256, rowbase, sA);
    __syncthreads();

    int wid = t >> 5, lane = t & 31;
    int hi = lane >> 4, ln = lane & 15;
    int n0 = wid * 32;
    int ntA = wid * 2, ntB = wid * 2 + 1;

    v8f acc0 = {}, acc1 = {};
#pragma unroll
    for (int ks = 0; ks < 8; ++ks) {
        v16bf a  = frag_ld(sA + (ks * 32 + lane) * 16);
        v16bf b0 = frag_ld(Bh1 + ((size_t)(ntA * 8 + ks) * 32 + lane) * 16);
        v16bf b1 = frag_ld(Bh1 + ((size_t)(ntB * 8 + ks) * 32 + lane) * 16);
        acc0 = wmma_bf16(a, b0, acc0);
        acc1 = wmma_bf16(a, b1, acc1);
    }

    FragC c0, c1; c0.v = acc0; c1.v = acc1;
#pragma unroll
    for (int j = 0; j < 8; ++j) {
        int row = j + (hi << 3);
        {
            int col = n0 + ln;
            float v = c0.f[j] + bh1[col];
            v = v > 0.0f ? v : 0.0f;
            atomicAdd(&part[row * NHH + (col >> 6)], v * Wh2[col]);
        }
        {
            int col = n0 + 16 + ln;
            float v = c1.f[j] + bh1[col];
            v = v > 0.0f ? v : 0.0f;
            atomicAdd(&part[row * NHH + (col >> 6)], v * Wh2[col]);
        }
    }
    __syncthreads();

    if (t < 16) {
        float s = 0.0f;
#pragma unroll
        for (int h = 0; h < NHH; ++h) {
            float z = part[t * NHH + h] + bh2[h];
            float a = 1.0f / (1.0f + __expf(-z));
            s += a * Wf[h];
        }
        attn[t] = s;
    }
    __syncthreads();

    for (int idx = t; idx < 16 * DD; idx += 256) {
        int r = idx >> 8, c = idx & 255;
        heF[(size_t)(rowbase + r) * DD + c] *= attn[r];
    }
}

// ---------------------------------------------------------------------------
// C(Mx256) = A(Mx256,f32) @ B(256x256, bf16 fragments) + bias
// ---------------------------------------------------------------------------
__global__ void gemm256x256_bias(const float* __restrict__ A,
                                 const unsigned short* __restrict__ B,
                                 const float* __restrict__ bias,
                                 float* __restrict__ C) {
    __shared__ __align__(32) unsigned short sA[8 * 512];   // 8 KB

    int t = threadIdx.x;
    int rowbase = blockIdx.x * 16;
    stage_a_tile(A, 256, rowbase, sA);
    __syncthreads();

    int wid = t >> 5, lane = t & 31;
    int hi = lane >> 4, ln = lane & 15;
    int n0 = wid * 32;
    int ntA = wid * 2, ntB = wid * 2 + 1;

    v8f acc0 = {}, acc1 = {};
#pragma unroll
    for (int ks = 0; ks < 8; ++ks) {
        v16bf a  = frag_ld(sA + (ks * 32 + lane) * 16);
        v16bf b0 = frag_ld(B + ((size_t)(ntA * 8 + ks) * 32 + lane) * 16);
        v16bf b1 = frag_ld(B + ((size_t)(ntB * 8 + ks) * 32 + lane) * 16);
        acc0 = wmma_bf16(a, b0, acc0);
        acc1 = wmma_bf16(a, b1, acc1);
    }

    FragC c0, c1; c0.v = acc0; c1.v = acc1;
#pragma unroll
    for (int j = 0; j < 8; ++j) {
        int row = rowbase + j + (hi << 3);
        int colA = n0 + ln;
        int colB = n0 + 16 + ln;
        C[(size_t)row * 256 + colA] = c0.f[j] + bias[colA];
        C[(size_t)row * 256 + colB] = c1.f[j] + bias[colB];
    }
}

// ---------------------------------------------------------------------------
// Fused MLP head, 16 rows per block:
// x1 = relu([self,cluster] @ Bfa1 + bfa1)   (512 -> 256)  -> LDS (frag order)
// x2 = relu(x1 @ Bfa2 + bfa2)               (256 -> 128)  -> LDS f32
// fw = softmax(x2 @ Wfa3^T + bfa3)          (128 -> 2)
// out = relu(self*fw0 + cluster*fw1 + feat)
// ---------------------------------------------------------------------------
__global__ void fuse_head(const float* __restrict__ selfF,
                          const float* __restrict__ clusterT,
                          const float* __restrict__ feat,
                          const unsigned short* __restrict__ Bfa1,
                          const float* __restrict__ bfa1,
                          const unsigned short* __restrict__ Bfa2,
                          const float* __restrict__ bfa2,
                          const float* __restrict__ Wfa3,
                          const float* __restrict__ bfa3,
                          float* __restrict__ out) {
    __shared__ __align__(32) unsigned short sA1[16 * 512]; // 16 KB (K=512)
    __shared__ __align__(32) unsigned short sX1[8 * 512];  //  8 KB (K=256, frag order)
    __shared__ float sX2[16 * 128];                        //  8 KB
    __shared__ float sFW[16 * 2];

    int t = threadIdx.x;
    int rowbase = blockIdx.x * 16;
    int wid = t >> 5, lane = t & 31;
    int hi = lane >> 4, ln = lane & 15;

    // stage concat([self, cluster]) as fragment-ordered bf16, K = 512
    for (int pi = t; pi < 16 * 32; pi += 256) {
        int plane = pi & 31;
        int ks    = pi >> 5;
        int row   = rowbase + (plane & 15);
        int phi   = plane >> 4;
        FragAB fa;
#pragma unroll
        for (int e = 0; e < 16; ++e) {
            int k = (ks << 5) + kmap(e, phi);
            float v = (k < 256) ? selfF[(size_t)row * 256 + k]
                                : clusterT[(size_t)row * 256 + (k - 256)];
            fa.u[e] = f2bfu(v);
        }
        *(v16bf*)(sA1 + pi * 16) = fa.v;
    }
    __syncthreads();

    // layer 1: K=512, N=256 (2 tiles per wave); write x1 in layer-2 frag order
    {
        int n0 = wid * 32;
        int ntA = wid * 2, ntB = wid * 2 + 1;
        v8f acc0 = {}, acc1 = {};
#pragma unroll
        for (int ks = 0; ks < 16; ++ks) {
            v16bf a  = frag_ld(sA1 + (ks * 32 + lane) * 16);
            v16bf b0 = frag_ld(Bfa1 + ((size_t)(ntA * 16 + ks) * 32 + lane) * 16);
            v16bf b1 = frag_ld(Bfa1 + ((size_t)(ntB * 16 + ks) * 32 + lane) * 16);
            acc0 = wmma_bf16(a, b0, acc0);
            acc1 = wmma_bf16(a, b1, acc1);
        }
        FragC c0, c1; c0.v = acc0; c1.v = acc1;
#pragma unroll
        for (int j = 0; j < 8; ++j) {
            int row = j + (hi << 3);
#pragma unroll
            for (int half = 0; half < 2; ++half) {
                int col = n0 + half * 16 + ln;
                float v = (half ? c1.f[j] : c0.f[j]) + bfa1[col];
                v = v > 0.0f ? v : 0.0f;
                int ks2, hi2, e2;
                ks2 = col >> 5;
                kmap_inv(col & 31, hi2, e2);
                int lane2 = (hi2 << 4) | row;
                sX1[(ks2 * 32 + lane2) * 16 + e2] = f2bfu(v);
            }
        }
    }
    __syncthreads();

    // layer 2: K=256, N=128 (1 tile per wave)
    {
        int nt = wid;
        v8f acc = {};
#pragma unroll
        for (int ks = 0; ks < 8; ++ks) {
            v16bf a = frag_ld(sX1 + (ks * 32 + lane) * 16);
            v16bf b = frag_ld(Bfa2 + ((size_t)(nt * 8 + ks) * 32 + lane) * 16);
            acc = wmma_bf16(a, b, acc);
        }
        FragC c; c.v = acc;
#pragma unroll
        for (int j = 0; j < 8; ++j) {
            int row = j + (hi << 3);
            int col = nt * 16 + ln;
            float v = c.f[j] + bfa2[col];
            sX2[row * 128 + col] = v > 0.0f ? v : 0.0f;
        }
    }
    __syncthreads();

    // layer 3 + softmax (2 logits per row)
    if (t < 16) {
        float l0 = bfa3[0], l1 = bfa3[1];
        for (int j = 0; j < 128; ++j) {
            float x = sX2[t * 128 + j];
            l0 += x * Wfa3[j];
            l1 += x * Wfa3[128 + j];
        }
        float m = fmaxf(l0, l1);
        float e0 = __expf(l0 - m), e1 = __expf(l1 - m);
        float inv = 1.0f / (e0 + e1);
        sFW[t * 2 + 0] = e0 * inv;
        sFW[t * 2 + 1] = e1 * inv;
    }
    __syncthreads();

    // fused output (residual + relu); re-read f32 self/cluster from global.
    // out aliases selfF: each element is read then written by the same thread.
    for (int idx = t; idx < 16 * 256; idx += 256) {
        int r = idx >> 8, c = idx & 255;
        size_t g = (size_t)(rowbase + r) * 256 + c;
        float v = selfF[g] * sFW[r * 2 + 0] + clusterT[g] * sFW[r * 2 + 1] + feat[g];
        out[g] = fmaxf(v, 0.0f);
    }
}

// ---------------------------------------------------------------------------
// Host-side orchestration
// ---------------------------------------------------------------------------
extern "C" void kernel_launch(void* const* d_in, const int* in_sizes, int n_in,
                              void* d_out, int out_size, void* d_ws, size_t ws_size,
                              hipStream_t stream) {
    (void)in_sizes; (void)n_in; (void)out_size; (void)ws_size;

    const float* feat     = (const float*)d_in[0];
    const int*   node_idx = (const int*)  d_in[1];
    const int*   he_idx   = (const int*)  d_in[2];
    const float* ew       = (const float*)d_in[3];
    const float* Ws       = (const float*)d_in[4];
    const float* bs       = (const float*)d_in[5];
    const float* Wc       = (const float*)d_in[6];
    const float* bc       = (const float*)d_in[7];
    const float* Wh1      = (const float*)d_in[8];
    const float* bh1      = (const float*)d_in[9];
    const float* Wh2      = (const float*)d_in[10];
    const float* bh2      = (const float*)d_in[11];
    const float* Wf       = (const float*)d_in[12];
    const float* Wfa1     = (const float*)d_in[13];
    const float* bfa1     = (const float*)d_in[14];
    const float* Wfa2     = (const float*)d_in[15];
    const float* bfa2     = (const float*)d_in[16];
    const float* Wfa3     = (const float*)d_in[17];
    const float* bfa3     = (const float*)d_in[18];

    // workspace carve-up (all offsets 32-byte aligned)
    char* base = (char*)d_ws;
    size_t off = 0;
    float* heF      = (float*)(base + off); off += (size_t)N_HE * DD * 4;   // 20.48 MB
    float* cluster  = (float*)(base + off); off += (size_t)N_P  * DD * 4;   // 102.4 MB
    float* clusterT = (float*)(base + off); off += (size_t)N_P  * DD * 4;   // 102.4 MB
    unsigned short* Bh1  = (unsigned short*)(base + off); off += 256 * 256 * 2;
    unsigned short* Bs   = (unsigned short*)(base + off); off += 256 * 256 * 2;
    unsigned short* Bc   = (unsigned short*)(base + off); off += 256 * 256 * 2;
    unsigned short* Bfa1 = (unsigned short*)(base + off); off += 512 * 256 * 2;
    unsigned short* Bfa2 = (unsigned short*)(base + off); off += 256 * 128 * 2;

    // zero segment-sum accumulators (graph-capturable memset nodes)
    hipMemsetAsync(heF,     0, (size_t)N_HE * DD * 4, stream);
    hipMemsetAsync(cluster, 0, (size_t)N_P  * DD * 4, stream);

    // weight prep -> fragment-ordered bf16 (tiny)
    swizzle_wh1<<<16, 256, 0, stream>>>(Wh1, Bh1);
    swizzle_wT <<<16, 256, 0, stream>>>(Ws,   Bs,   256, 256);
    swizzle_wT <<<16, 256, 0, stream>>>(Wc,   Bc,   256, 256);
    swizzle_wT <<<32, 256, 0, stream>>>(Wfa1, Bfa1, 256, 512);
    swizzle_wT <<< 8, 256, 0, stream>>>(Wfa2, Bfa2, 128, 256);

    // he_feat = segment_sum(feat[node_idx] * ew, he_idx)
    scatter_weighted_add<<<(N_INC + 3) / 4, 256, 0, stream>>>(
        feat, node_idx, he_idx, ew, heF, N_INC);

    // hyperedge attention + in-place reweight of he_feat
    he_attn_kernel<<<N_HE / 16, 256, 0, stream>>>(heF, Bh1, bh1, Wh2, bh2, Wf);

    // cluster = segment_sum(he_weighted[he_idx] * ew, node_idx)
    scatter_weighted_add<<<(N_INC + 3) / 4, 256, 0, stream>>>(
        heF, he_idx, node_idx, ew, cluster, N_INC);

    // self_feat (stored in d_out as scratch) and cluster_t
    gemm256x256_bias<<<N_P / 16, 256, 0, stream>>>(feat,    Bs, bs, (float*)d_out);
    gemm256x256_bias<<<N_P / 16, 256, 0, stream>>>(cluster, Bc, bc, clusterT);

    // fused MLP head + residual + relu -> d_out
    fuse_head<<<N_P / 16, 256, 0, stream>>>(
        (const float*)d_out, clusterT, feat,
        Bfa1, bfa1, Bfa2, bfa2, Wfa3, bfa3, (float*)d_out);
}
